// MultiHeadAttention_35476429865961
// MI455X (gfx1250) — matrix operand order repaired and checked by reading, not compiled, not run on hardware
//
#include <hip/hip_runtime.h>
#include <stdint.h>

// ---- shapes ----
#define DIM   1024
#define HEADS 16
#define HD    64
#define SEQ   2048
#define BATCH 2
#define ROWS  (BATCH * SEQ)          // 4096
#define SCALE (1.0f / 64.0f)         // 1/head_dim per reference

// ---- CDNA5 WMMA vector types ----
typedef __attribute__((ext_vector_type(16))) __bf16    v16bf;
typedef __attribute__((ext_vector_type(8)))  float     v8f;
typedef __attribute__((ext_vector_type(8)))  uint16_t  v8u;
typedef __attribute__((ext_vector_type(16))) uint16_t  v16u;
typedef __attribute__((ext_vector_type(4)))  uint32_t  u32x4;
typedef __attribute__((ext_vector_type(8)))  uint32_t  u32x8;

__device__ __forceinline__ uint16_t f2bf(float f) {
  uint32_t u = __builtin_bit_cast(uint32_t, f);
  uint32_t r = u + 0x7FFFu + ((u >> 16) & 1u);   // round-to-nearest-even
  return (uint16_t)(r >> 16);
}

__device__ __forceinline__ v8f wmma_bf16(v16u a_bits, v16u b_bits, v8f c) {
  v16bf a = __builtin_bit_cast(v16bf, a_bits);
  v16bf b = __builtin_bit_cast(v16bf, b_bits);
  return __builtin_amdgcn_wmma_f32_16x16x32_bf16(false, a, false, b, (short)0, c,
                                                 false, false);
}

// A-fragment: lane holds row M=lane%16; K runs [16h,16h+16) contiguous
__device__ __forceinline__ v16u afrag(const uint16_t* base_row, int half) {
  v8u lo = *(const v8u*)(base_row + 16 * half);
  v8u hi = *(const v8u*)(base_row + 16 * half + 8);
  return __builtin_shufflevector(lo, hi, 0, 1, 2, 3, 4, 5, 6, 7,
                                         8, 9, 10, 11, 12, 13, 14, 15);
}

// ---- Tensor Data Mover: DMA a 2D bf16 tile [tile1 rows x tile0 cols] from
// global (row pitch = stride0 elements) into LDS at lds_off, rows packed.
// D# layout per CDNA5 ISA (group0 + group1, 2D form,
// VADDR2/3 = NULL). tensor_dim == tile_dim so the OOB clamp never triggers.
__device__ __forceinline__ void tdm_load_2d(uint32_t lds_off, const void* gaddr,
                                            uint32_t tile0, uint32_t tile1,
                                            uint32_t stride0) {
  uint64_t ga = (uint64_t)(uintptr_t)gaddr;
  u32x4 g0;
  g0[0] = 1u;                                    // count=1, user descriptor
  g0[1] = lds_off;                               // lds_addr
  g0[2] = (uint32_t)ga;                          // global_addr[31:0]
  g0[3] = ((uint32_t)(ga >> 32) & 0x01FFFFFFu) | 0x80000000u;  // addr[56:32]|type=2
  u32x8 g1;
  g1[0] = (1u << 16);                            // workgroup_mask=0, data_size=2B
  g1[1] = (tile0 & 0xFFFFu) << 16;               // tensor_dim0[15:0]
  g1[2] = (tile1 & 0xFFFFu) << 16;               // tensor_dim0[31:16]=0, dim1[15:0]
  g1[3] = (tile0 & 0xFFFFu) << 16;               // dim1[31:16]=0, tile_dim0
  g1[4] = tile1 & 0xFFFFu;                       // tile_dim1, tile_dim2=0
  g1[5] = stride0;                               // tensor_dim0_stride[31:0]
  g1[6] = 0u;                                    // stride0[47:32], dim1_stride lo
  g1[7] = 0u;
  asm volatile("tensor_load_to_lds %0, %1" ::"s"(g0), "s"(g1) : "memory");
}

__device__ __forceinline__ uint32_t lds_offset(const void* p) {
  // generic (flat) shared pointers carry the LDS byte offset in the low 32 bits
  return (uint32_t)(uintptr_t)p;
}

// ---------------- kernel 1: fp32 -> bf16 ----------------
__global__ void cvt_bf16(const float* __restrict__ in, uint16_t* __restrict__ out,
                         int n) {
  int i = blockIdx.x * blockDim.x + threadIdx.x;
  int stride = gridDim.x * blockDim.x;
  for (; i < n; i += stride) out[i] = f2bf(in[i]);
}

// ---------------- kernels 2/4: GEMM  y = A(bf16) @ W(bf16) + bias ----------
// MODE 0: scatter bf16 into [B,H,S,hd] (QKV).  MODE 1: fp32 row-major [M,DIM].
template <int MODE>
__global__ __launch_bounds__(256) void gemm_proj(
    const uint16_t* __restrict__ A,    // [ROWS][DIM] bf16
    const uint16_t* __restrict__ W,    // [DIM][DIM]  bf16 (in,out)
    const float* __restrict__ bias,    // [DIM] fp32
    uint16_t* __restrict__ out_bf,     // MODE 0
    float* __restrict__ out_f) {       // MODE 1
  __shared__ __align__(128) uint16_t As[128][64];  // [m][k]  16 KB (TDM-filled)
  __shared__ __align__(128) uint16_t Bt[64][64];   // [n][k]   8 KB (transposed)

  const int tid = threadIdx.x;
  const int wave = tid >> 5, lane = tid & 31;
  const int half = lane >> 4, ln = lane & 15;
  const int wm = wave >> 1, wn = wave & 1;         // 4x2 wave grid
  const int mb = blockIdx.y * 128, nb = blockIdx.x * 64;

  v8f acc[2][2] = {};

  for (int kk = 0; kk < DIM; kk += 64) {
    // A tile 128x64 via Tensor Data Mover (wave 0 only; TDM ignores EXEC)
    if (wave == 0)
      tdm_load_2d(lds_offset(&As[0][0]), A + (size_t)mb * DIM + kk,
                  /*tile0=*/64, /*tile1=*/128, /*stride0=*/DIM);
    {  // W tile 64x64, store transposed [n][k]
      int k = tid >> 2, seg = (tid & 3) * 16;
#pragma unroll
      for (int s2 = 0; s2 < 2; ++s2) {
        v8u v = *(const v8u*)(W + (size_t)(kk + k) * DIM + nb + seg + s2 * 8);
#pragma unroll
        for (int e = 0; e < 8; ++e) Bt[seg + s2 * 8 + e][k] = v[e];
      }
    }
    if (wave == 0) __builtin_amdgcn_s_wait_tensorcnt(0);
    __syncthreads();

#pragma unroll
    for (int c = 0; c < 2; ++c) {                  // two K=32 windows
      v16u af[2], bfv[2];
#pragma unroll
      for (int mt = 0; mt < 2; ++mt)
        af[mt] = afrag(&As[wm * 32 + mt * 16 + ln][c * 32], half);
#pragma unroll
      for (int nt = 0; nt < 2; ++nt)               // B: contiguous 16 at K=16*half
        bfv[nt] = *(const v16u*)&Bt[wn * 32 + nt * 16 + ln][c * 32 + 16 * half];
#pragma unroll
      for (int mt = 0; mt < 2; ++mt)
#pragma unroll
        for (int nt = 0; nt < 2; ++nt)
          acc[mt][nt] = wmma_bf16(af[mt], bfv[nt], acc[mt][nt]);
    }
    __syncthreads();
  }

#pragma unroll
  for (int mt = 0; mt < 2; ++mt)
#pragma unroll
    for (int nt = 0; nt < 2; ++nt) {
      int N = nb + wn * 32 + nt * 16 + ln;
      float bv = bias[N];
#pragma unroll
      for (int r = 0; r < 8; ++r) {
        int Mg = mb + wm * 32 + mt * 16 + r + 8 * half;
        float v = acc[mt][nt][r] + bv;
        if (MODE == 0) {
          int b = Mg >> 11, s = Mg & (SEQ - 1);
          int h = N >> 6, d = N & (HD - 1);
          out_bf[(((size_t)(b * HEADS + h)) * SEQ + s) * HD + d] = f2bf(v);
        } else {
          out_f[(size_t)Mg * DIM + N] = v;
        }
      }
    }
}

// ---------------- kernel 3: flash attention (mask keeps j >= i) ------------
__global__ __launch_bounds__(256) void attention_kernel(
    const uint16_t* __restrict__ Q,   // [B*H][S][HD] bf16
    const uint16_t* __restrict__ K,
    const uint16_t* __restrict__ V,
    uint16_t* __restrict__ O) {       // [ROWS][DIM] bf16 (heads merged)
  __shared__ __align__(128) uint16_t Ks[128][64];       // [key][d]  16 KB (TDM)
  __shared__ __align__(128) uint16_t Vt[64][128];       // [d][key]  16 KB
  __shared__ __align__(128) uint16_t Ps[8][16][128];    // per-wave P 32 KB

  const int tid = threadIdx.x;
  const int wave = tid >> 5, lane = tid & 31;
  const int half = lane >> 4, ln = lane & 15;

  const int qb = blockIdx.x * 128;
  const int bh = blockIdx.y;
  const int b = bh >> 4, h = bh & 15;

  const uint16_t* Qbh = Q + (size_t)bh * SEQ * HD;
  const uint16_t* Kbh = K + (size_t)bh * SEQ * HD;
  const uint16_t* Vbh = V + (size_t)bh * SEQ * HD;

  // per-wave Q fragments: 16 rows x 64, two K=32 windows (kept in VGPRs)
  v16u qf[2];
  {
    const uint16_t* qp = Qbh + (size_t)(qb + wave * 16 + ln) * HD;
#pragma unroll
    for (int c = 0; c < 2; ++c) qf[c] = afrag(qp + c * 32, half);
  }

  v8f o[4] = {};
  float m_r[8], l_r[8];
#pragma unroll
  for (int r = 0; r < 8; ++r) { m_r[r] = -1.0e30f; l_r[r] = 0.0f; }

  // mask keeps j >= i -> only key blocks at/after the q block contribute
  for (int kb = qb; kb < SEQ; kb += 128) {
    // K tile 128x64 via Tensor Data Mover (wave 0 only)
    if (wave == 0)
      tdm_load_2d(lds_offset(&Ks[0][0]), Kbh + (size_t)kb * HD,
                  /*tile0=*/64, /*tile1=*/128, /*stride0=*/HD);
    {  // stage V transposed [d][key]
      int key = tid >> 1, dh = (tid & 1) * 32;
      const uint16_t* src = Vbh + (size_t)(kb + key) * HD + dh;
#pragma unroll
      for (int q4 = 0; q4 < 4; ++q4) {
        v8u v = *(const v8u*)(src + q4 * 8);
#pragma unroll
        for (int e = 0; e < 8; ++e) Vt[dh + q4 * 8 + e][key] = v[e];
      }
    }
    if (wave == 0) __builtin_amdgcn_s_wait_tensorcnt(0);
    __syncthreads();

    // prefetch next key block's K/V tiles into cache (global_prefetch_b8)
    if (kb + 128 < SEQ) {
      __builtin_prefetch(Kbh + (size_t)(kb + 128) * HD + tid * 32, 0, 1);
      __builtin_prefetch(Vbh + (size_t)(kb + 128) * HD + tid * 32, 0, 1);
    }

    // S = Q K^T : 8 key-tiles of 16, contraction hd=64 (2 WMMA each)
    v8f s[8];
#pragma unroll
    for (int t = 0; t < 8; ++t) {
      v8f a = {};
#pragma unroll
      for (int c = 0; c < 2; ++c) {
        v16u kf = *(const v16u*)&Ks[t * 16 + ln][c * 32 + 16 * half];
        a = wmma_bf16(qf[c], kf, a);
      }
      s[t] = a;
    }

    // scale + mask (only first key block is partial)
#pragma unroll
    for (int t = 0; t < 8; ++t)
#pragma unroll
      for (int r = 0; r < 8; ++r) {
        float x = s[t][r] * SCALE;
        if (kb == qb) {
          int i = wave * 16 + r + 8 * half;  // local q row
          int j = t * 16 + ln;               // local key
          if (j < i) x = -1.0e30f;
        }
        s[t][r] = x;
      }

    // online softmax: row stats live per (r, half) replicated across 16 lanes
    float alpha[8];
#pragma unroll
    for (int r = 0; r < 8; ++r) {
      float mx = s[0][r];
#pragma unroll
      for (int t = 1; t < 8; ++t) mx = fmaxf(mx, s[t][r]);
#pragma unroll
      for (int off = 1; off < 16; off <<= 1) mx = fmaxf(mx, __shfl_xor(mx, off, 32));
      float mnew = fmaxf(m_r[r], mx);
      alpha[r] = __expf(m_r[r] - mnew);
      m_r[r] = mnew;
    }
#pragma unroll
    for (int r = 0; r < 8; ++r) {
      float sum = 0.0f;
#pragma unroll
      for (int t = 0; t < 8; ++t) {
        float p = __expf(s[t][r] - m_r[r]);
        s[t][r] = p;
        sum += p;
      }
#pragma unroll
      for (int off = 1; off < 16; off <<= 1) sum += __shfl_xor(sum, off, 32);
      l_r[r] = l_r[r] * alpha[r] + sum;
#pragma unroll
      for (int n = 0; n < 4; ++n) o[n][r] = o[n][r] * alpha[r];
    }

    // C-layout -> A-layout via per-wave LDS (in-order DS within a wave)
#pragma unroll
    for (int t = 0; t < 8; ++t)
#pragma unroll
      for (int r = 0; r < 8; ++r)
        Ps[wave][r + 8 * half][t * 16 + ln] = f2bf(s[t][r]);

    // O += P V : contraction 128 keys (4 WMMA) x 4 hd-tiles
#pragma unroll
    for (int kk = 0; kk < 4; ++kk) {
      v16u pf = afrag(&Ps[wave][ln][kk * 32], half);
#pragma unroll
      for (int n = 0; n < 4; ++n) {
        v16u vf = *(const v16u*)&Vt[n * 16 + ln][kk * 32 + 16 * half];
        o[n] = wmma_bf16(pf, vf, o[n]);
      }
    }
    __syncthreads();
  }

  // normalize and merge heads: out[b*S + s][h*64 + d]
#pragma unroll
  for (int r = 0; r < 8; ++r) {
    float inv = 1.0f / l_r[r];
    int row = qb + wave * 16 + r + 8 * half;
    size_t orow = (size_t)(b * SEQ + row);
#pragma unroll
    for (int n = 0; n < 4; ++n) {
      int col = h * HD + n * 16 + ln;
      O[orow * DIM + col] = f2bf(o[n][r] * inv);
    }
  }
}

// ---------------- host launch ----------------
extern "C" void kernel_launch(void* const* d_in, const int* in_sizes, int n_in,
                              void* d_out, int out_size, void* d_ws, size_t ws_size,
                              hipStream_t stream) {
  (void)in_sizes; (void)n_in; (void)out_size; (void)ws_size;
  const float* x  = (const float*)d_in[0];
  const float* wq = (const float*)d_in[1];
  const float* bq = (const float*)d_in[2];
  const float* wk = (const float*)d_in[3];
  const float* bk = (const float*)d_in[4];
  const float* wv = (const float*)d_in[5];
  const float* bv = (const float*)d_in[6];
  const float* wo = (const float*)d_in[7];
  const float* bo = (const float*)d_in[8];

  const size_t SZ_X = (size_t)ROWS * DIM;   // 4 Mi elems
  const size_t SZ_W = (size_t)DIM * DIM;    // 1 Mi elems

  uint16_t* xb    = (uint16_t*)d_ws;
  uint16_t* wqb   = xb + SZ_X;
  uint16_t* wkb   = wqb + SZ_W;
  uint16_t* wvb   = wkb + SZ_W;
  uint16_t* wob   = wvb + SZ_W;
  uint16_t* Qb    = wob + SZ_W;
  uint16_t* Kb    = Qb + SZ_X;
  uint16_t* Vb    = Kb + SZ_X;
  uint16_t* attnb = Vb + SZ_X;   // total 48 MB of workspace

  cvt_bf16<<<2048, 256, 0, stream>>>(x,  xb,  (int)SZ_X);
  cvt_bf16<<<1024, 256, 0, stream>>>(wq, wqb, (int)SZ_W);
  cvt_bf16<<<1024, 256, 0, stream>>>(wk, wkb, (int)SZ_W);
  cvt_bf16<<<1024, 256, 0, stream>>>(wv, wvb, (int)SZ_W);
  cvt_bf16<<<1024, 256, 0, stream>>>(wo, wob, (int)SZ_W);

  dim3 gg(DIM / 64, ROWS / 128);
  gemm_proj<0><<<gg, 256, 0, stream>>>(xb, wqb, bq, Qb, nullptr);
  gemm_proj<0><<<gg, 256, 0, stream>>>(xb, wkb, bk, Kb, nullptr);
  gemm_proj<0><<<gg, 256, 0, stream>>>(xb, wvb, bv, Vb, nullptr);

  attention_kernel<<<dim3(SEQ / 128, BATCH * HEADS), 256, 0, stream>>>(Qb, Kb, Vb,
                                                                       attnb);

  gemm_proj<1><<<gg, 256, 0, stream>>>(attnb, wob, bo, nullptr, (float*)d_out);
}
